// BahdanauAttention_51582557225637
// MI455X (gfx1250) — matrix operand order repaired and checked by reading, not compiled, not run on hardware
//
#include <hip/hip_runtime.h>
#include <hip/hip_bf16.h>
#include <math.h>

#define N_B   32
#define T_DIM 2048
#define E_DIM 1024
#define U_DIM 1024

typedef __attribute__((ext_vector_type(16))) __bf16 v16bf;
typedef __attribute__((ext_vector_type(8)))  float  v8f;

union AFrag { unsigned int u[8]; v16bf v; };

__device__ __forceinline__ unsigned int pack_bf16x2(float a, float b) {
    union { float f; unsigned int u; } ca, cb;
    ca.f = a; cb.f = b;
    unsigned int ra = (ca.u + 0x7FFFu + ((ca.u >> 16) & 1u)) >> 16;
    unsigned int rb = (cb.u + 0x7FFFu + ((cb.u >> 16) & 1u)) >> 16;
    return (ra & 0xFFFFu) | (rb << 16);
}

// branch-free tanh: 1 - 2/(exp(2x)+1); saturates correctly at +/-1
__device__ __forceinline__ float fast_tanh(float x) {
    float ex = __expf(2.0f * x);                  // v_exp_f32 (TRANS, co-executes)
    return 1.0f - 2.0f * __builtin_amdgcn_rcpf(ex + 1.0f);
}

// ---------------------------------------------------------------- Wk fp32 -> bf16
__global__ __launch_bounds__(256) void wk_cvt_kernel(const float* __restrict__ wk,
                                                     unsigned int* __restrict__ out) {
    int i = blockIdx.x * 256 + threadIdx.x;          // float4 index; total U*E/4
    float4 f = ((const float4*)wk)[i];
    uint2 p;
    p.x = pack_bf16x2(f.x, f.y);
    p.y = pack_bf16x2(f.z, f.w);
    ((uint2*)out)[i] = p;
}

// ---------------------------------------------------------------- proj_query GEMV
__global__ __launch_bounds__(256) void pq_kernel(const float* __restrict__ q,
                                                 const float* __restrict__ Wq,
                                                 float* __restrict__ pq) {
    __shared__ float qs[E_DIM];
    int n = blockIdx.x;
    for (int i = threadIdx.x; i < E_DIM; i += 256) qs[i] = q[(size_t)n * E_DIM + i];
    __syncthreads();
    for (int u = threadIdx.x; u < U_DIM; u += 256) {
        const float4* w  = (const float4*)(Wq + (size_t)u * E_DIM);
        const float4* qv = (const float4*)qs;
        float s = 0.f;
        #pragma unroll 4
        for (int d = 0; d < E_DIM / 4; ++d) {
            float4 a = w[d], b = qv[d];
            s += a.x * b.x + a.y * b.y + a.z * b.z + a.w * b.w;
        }
        pq[(size_t)n * U_DIM + u] = s;
    }
}

// ---------------------------------------------------------------- fused WMMA logits
// grid = (T/64, N), block = 128 (4 waves). Each wave owns one 16-row t-tile and
// iterates the FULL u range, so it produces completely reduced weights for its rows.
// All 4 waves fetch identical B (Wk) fragments in lockstep -> served from WGP$,
// cutting Wk L2 traffic 4x vs. a 16-row block.
// weights[n,t] = sum_u v[u] * tanh(pq[n,u] + (enc[n,t,:] . Wk[u,:]))
__global__ __launch_bounds__(128) void weights_kernel(const float* __restrict__ enc,
                                                      const unsigned short* __restrict__ wk_bf,
                                                      const float* __restrict__ pq,
                                                      const float* __restrict__ v,
                                                      float* __restrict__ weights) {
    __shared__ unsigned int As[64 * (E_DIM / 2)];   // 64 x 1024 bf16 = 128 KB
    __shared__ float pqs[U_DIM];                    // 4 KB
    __shared__ float vs[U_DIM];                     // 4 KB

    const int n    = blockIdx.y;
    const int t0   = blockIdx.x * 64;
    const int tid  = threadIdx.x;
    const int wave = tid >> 5;
    const int lane = tid & 31;
    const int m    = lane & 15;       // A row / C column index
    const int half = lane >> 4;       // K-half selector

    // ---- stage enc[n, t0:t0+64, :] into LDS as bf16; stage pq row and v
    const float* src = enc + ((size_t)n * T_DIM + t0) * E_DIM;
    for (int i = tid; i < 64 * E_DIM / 4; i += 128) {   // 16384 float4s
        float4 f = ((const float4*)src)[i];
        As[i * 2 + 0] = pack_bf16x2(f.x, f.y);
        As[i * 2 + 1] = pack_bf16x2(f.z, f.w);
    }
    for (int i = tid; i < U_DIM; i += 128) {
        pqs[i] = pq[(size_t)n * U_DIM + i];
        vs[i]  = v[i];
    }
    __syncthreads();

    float wsum[8];
    #pragma unroll
    for (int r = 0; r < 8; ++r) wsum[r] = 0.f;

    // this wave's A rows live at LDS rows [wave*16, wave*16+16)
    const unsigned int* rowp = &As[(wave * 16 + m) * (E_DIM / 2)];

    for (int ut = 0; ut < U_DIM / 16; ++ut) {
        __syncthreads();                              // keep waves lockstep -> WGP$ hits on B
        const int ucol  = ut * 16 + m;                // this lane's C column
        const float vv  = vs[ucol];
        const float pqv = pqs[ucol];
        const unsigned int* wrow =
            (const unsigned int*)(wk_bf + (size_t)ucol * E_DIM);

        v8f acc = {0.f, 0.f, 0.f, 0.f, 0.f, 0.f, 0.f, 0.f};

        for (int e0 = 0; e0 < E_DIM; e0 += 32) {
            // A fragment: lane<16 -> K {e0..e0+7, e0+16..e0+23}; lane>=16 -> +8
            AFrag a;
            const int kb = half * 8;
            uint4 x0 = *(const uint4*)&rowp[(e0 + kb) >> 1];
            uint4 x1 = *(const uint4*)&rowp[(e0 + 16 + kb) >> 1];
            a.u[0] = x0.x; a.u[1] = x0.y; a.u[2] = x0.z; a.u[3] = x0.w;
            a.u[4] = x1.x; a.u[5] = x1.y; a.u[6] = x1.z; a.u[7] = x1.w;

            // B fragment: lane<16 -> K {e0..e0+7, e0+16..e0+23}; lane>=16 -> +8
            AFrag b;
            const int ke = (e0 >> 1) + half * 4;      // uint index, 16B aligned
            uint4 y0 = *(const uint4*)&wrow[ke];
            uint4 y1 = *(const uint4*)&wrow[ke + 8];
            b.u[0] = y0.x; b.u[1] = y0.y; b.u[2] = y0.z; b.u[3] = y0.w;
            b.u[4] = y1.x; b.u[5] = y1.y; b.u[6] = y1.z; b.u[7] = y1.w;

            __builtin_prefetch(&wrow[ke + 32], 0, 0); // global_prefetch_b8

            acc = __builtin_amdgcn_wmma_f32_16x16x32_bf16(
                false, a.v, false, b.v, (short)0, acc, false, false);
        }

        // consume the tile in-register: rows r + 8*half, column m for this lane
        #pragma unroll
        for (int r = 0; r < 8; ++r)
            wsum[r] += vv * fast_tanh(pqv + acc[r]);
    }

    // reduce over the 16 columns within each half-wave; rows r + 8*half complete
    #pragma unroll
    for (int r = 0; r < 8; ++r) {
        float s = wsum[r];
        #pragma unroll
        for (int off = 1; off < 16; off <<= 1) s += __shfl_xor(s, off, 32);
        wsum[r] = s;
    }
    if (m == 0) {
        float* wout = weights + (size_t)n * T_DIM + t0 + wave * 16 + half * 8;
        #pragma unroll
        for (int r = 0; r < 8; ++r) wout[r] = wsum[r];
    }
}

// ---------------------------------------------------------------- masked softmax
__global__ __launch_bounds__(256) void softmax_kernel(const float* __restrict__ w,
                                                      const int* __restrict__ len,
                                                      float* __restrict__ alig) {
    __shared__ float sh[T_DIM];
    __shared__ float red[256];
    const int n = blockIdx.x;
    const int L = len[n];
    float mx = -INFINITY;
    for (int t = threadIdx.x; t < T_DIM; t += 256) {
        float x = (t < L) ? w[(size_t)n * T_DIM + t] : -INFINITY;
        sh[t] = x;
        mx = fmaxf(mx, x);
    }
    red[threadIdx.x] = mx; __syncthreads();
    for (int s = 128; s > 0; s >>= 1) {
        if (threadIdx.x < s) red[threadIdx.x] = fmaxf(red[threadIdx.x], red[threadIdx.x + s]);
        __syncthreads();
    }
    mx = red[0]; __syncthreads();
    float sum = 0.f;
    for (int t = threadIdx.x; t < T_DIM; t += 256) {
        float e = expf(sh[t] - mx);     // exp(-inf) = 0 for masked slots
        sh[t] = e; sum += e;
    }
    red[threadIdx.x] = sum; __syncthreads();
    for (int s = 128; s > 0; s >>= 1) {
        if (threadIdx.x < s) red[threadIdx.x] += red[threadIdx.x + s];
        __syncthreads();
    }
    const float inv = 1.0f / red[0];
    for (int t = threadIdx.x; t < T_DIM; t += 256)
        alig[(size_t)n * T_DIM + t] = sh[t] * inv;
}

// ---------------------------------------------------------------- contexts
// grid = (E/256, N), block = 256. ctx[n,e] = sum_t alpha[n,t] * enc[n,t,e]
__global__ __launch_bounds__(256) void ctx_kernel(const float* __restrict__ enc,
                                                  const float* __restrict__ alig,
                                                  float* __restrict__ ctx) {
    __shared__ float al[T_DIM];
    const int n = blockIdx.y;
    for (int t = threadIdx.x; t < T_DIM; t += 256) al[t] = alig[(size_t)n * T_DIM + t];
    __syncthreads();
    const int e = blockIdx.x * 256 + threadIdx.x;
    const float* base = enc + (size_t)n * T_DIM * E_DIM + e;
    float s = 0.f;
    #pragma unroll 4
    for (int t = 0; t < T_DIM; ++t) s += al[t] * base[(size_t)t * E_DIM];
    ctx[(size_t)n * E_DIM + e] = s;
}

// ---------------------------------------------------------------- host launcher
extern "C" void kernel_launch(void* const* d_in, const int* in_sizes, int n_in,
                              void* d_out, int out_size, void* d_ws, size_t ws_size,
                              hipStream_t stream) {
    const float* queries = (const float*)d_in[0];   // [N,1,D_DEC]
    const float* enc     = (const float*)d_in[1];   // [N,T,E]
    const int*   lengths = (const int*)  d_in[2];   // [N]
    const float* v       = (const float*)d_in[3];   // [U]
    const float* Wq      = (const float*)d_in[4];   // [U,D_DEC]
    const float* Wk      = (const float*)d_in[5];   // [U,E]

    float* out_ctx  = (float*)d_out;                         // [N,E]
    float* out_alig = (float*)d_out + (size_t)N_B * E_DIM;   // [N,T]

    // workspace layout (aligned): Wk bf16 (2 MB) | pq (128 KB) | weights (256 KB)
    char* ws = (char*)d_ws;
    unsigned int* wk_bf_u = (unsigned int*)ws;
    unsigned short* wk_bf = (unsigned short*)ws;
    float* pq      = (float*)(ws + (size_t)U_DIM * E_DIM * 2);
    float* weights = (float*)(ws + (size_t)U_DIM * E_DIM * 2 + (size_t)N_B * U_DIM * 4);

    wk_cvt_kernel<<<(U_DIM * E_DIM / 4) / 256, 256, 0, stream>>>(Wk, wk_bf_u);
    pq_kernel<<<N_B, 256, 0, stream>>>(queries, Wq, pq);

    dim3 gw(T_DIM / 64, N_B);
    weights_kernel<<<gw, 128, 0, stream>>>(enc, wk_bf, pq, v, weights);

    softmax_kernel<<<N_B, 256, 0, stream>>>(weights, lengths, out_alig);

    dim3 gc(E_DIM / 256, N_B);
    ctx_kernel<<<gc, 256, 0, stream>>>(enc, out_alig, out_ctx);
}
